// SonarNet_49959059587720
// MI455X (gfx1250) — compile-verified, matrix operand order checked by reading, not run on hardware
//
#include <hip/hip_runtime.h>
#include <math.h>

typedef __attribute__((ext_vector_type(16))) _Float16 v16h;
typedef __attribute__((ext_vector_type(8)))  _Float16 v8h;
typedef __attribute__((ext_vector_type(8)))  float    v8f;

#define LRELU(v) ((v) > 0.f ? (v) : 0.01f * (v))

// ---------------------------------------------------------------------------
// WMMA GEMM: C[rows,O] = act(A[rows,K] @ W[O,K]^T + bias[O]), all f16 operands
// (f32 accumulate, f16 output). Preconditions enforced by the launcher:
//   rows % 64 == 0, K % 32 == 0, O % 16 == 0, A/W zero-padded to K.
// 256 threads = 8 wave32s; each wave computes a 64x16 strip of C as 4 row
// tiles sharing one B fragment -> 4 independent v_wmma per K step, zero
// predication in the inner loop (EXEC stays all-1s as WMMA requires).
// Per-lane fragment = two 16B-aligned b128 loads (halves [k0+8g, +8) and
// [k0+16+8g, +8)) matching the ISA 16-bit A/B register layout.
// ---------------------------------------------------------------------------
__global__ void wmma_gemm_kernel(const _Float16* __restrict__ A,
                                 const _Float16* __restrict__ W,
                                 const float* __restrict__ bias,
                                 _Float16* __restrict__ C,
                                 int rows, int K, int O, int act)
{
  const int lane = threadIdx.x & 31;
  const int wave = threadIdx.x >> 5;
  const int colTiles = O >> 4;
  const int rowBlocks = rows >> 6;             // 64-row strips
  const int nTiles = rowBlocks * colTiles;
  const int tile = blockIdx.x * 8 + wave;
  if (tile >= nTiles) return;                  // wave-uniform
  const int tr = tile / colTiles;
  const int tc = tile - tr * colTiles;
  const int row0 = tr << 6, col0 = tc << 4;

  const int mrow = lane & 15;
  const int g = lane >> 4;

  const _Float16* A0 = A + (size_t)(row0 + mrow) * K;
  const _Float16* Wr = W + (size_t)(col0 + mrow) * K;
  const size_t rstep = (size_t)16 * K;

  v8f acc0 = {}, acc1 = {}, acc2 = {}, acc3 = {};
  for (int k0 = 0; k0 < K; k0 += 32) {
    const int o0 = k0 + g * 8;
    const int o1 = k0 + 16 + g * 8;
    const v8h bl = *(const v8h*)(Wr + o0);
    const v8h bh = *(const v8h*)(Wr + o1);
    const v16h b = __builtin_shufflevector(bl, bh, 0, 1, 2, 3, 4, 5, 6, 7,
                                           8, 9, 10, 11, 12, 13, 14, 15);
#define WMMA_STEP(ACC, T)                                                      \
    {                                                                          \
      const v8h al = *(const v8h*)(A0 + (size_t)(T) * rstep + o0);             \
      const v8h ah = *(const v8h*)(A0 + (size_t)(T) * rstep + o1);             \
      const v16h a = __builtin_shufflevector(al, ah, 0, 1, 2, 3, 4, 5, 6, 7,   \
                                             8, 9, 10, 11, 12, 13, 14, 15);    \
      ACC = __builtin_amdgcn_wmma_f32_16x16x32_f16(false, a, false, b,         \
                                                   (short)0, ACC, false, false); \
    }
    WMMA_STEP(acc0, 0)
    WMMA_STEP(acc1, 1)
    WMMA_STEP(acc2, 2)
    WMMA_STEP(acc3, 3)
#undef WMMA_STEP
  }

  const int n = col0 + mrow;
  const float bn = bias[n];
#pragma unroll
  for (int t4 = 0; t4 < 4; ++t4) {
    const v8f acc = (t4 == 0) ? acc0 : (t4 == 1) ? acc1 : (t4 == 2) ? acc2 : acc3;
    const int rbase = row0 + t4 * 16 + g * 8;  // D layout: VGPR r -> M = r + 8g
#pragma unroll
    for (int r = 0; r < 8; ++r) {
      float v = acc[r] + bn;
      if (act) v = LRELU(v);
      C[(size_t)(rbase + r) * O + n] = (_Float16)v;
    }
  }
}

// --------------- weight convert f32 -> f16 with K zero-padding --------------
__global__ void wcvt_kernel(const float* __restrict__ w, _Float16* __restrict__ out,
                            int O, int K, int Kp)
{
  const int i = blockIdx.x * blockDim.x + threadIdx.x;
  if (i >= O * Kp) return;
  const int o = i / Kp, k = i - o * Kp;
  out[i] = (k < K) ? (_Float16)w[(size_t)o * K + k] : (_Float16)0.f;
}

// ------------------------- encoder layer 1 (3 -> 64) ------------------------
__global__ void enc1_kernel(const float* __restrict__ x, const float* __restrict__ w,
                            const float* __restrict__ bias, _Float16* __restrict__ out,
                            int BN)
{
  const int t = blockIdx.x * blockDim.x + threadIdx.x;
  if (t >= BN * 64) return;
  const int r = t >> 6, c = t & 63;
  const float* xi = x + (size_t)r * 3;
  out[t] = (_Float16)(w[c * 3] * xi[0] + w[c * 3 + 1] * xi[1] + w[c * 3 + 2] * xi[2] + bias[c]);
}

// ----------------------- GroupNorm(1, C) over (C,N) -------------------------
__global__ void gn_stats_kernel(const _Float16* __restrict__ X, float* __restrict__ stats,
                                int perSample)
{
  __shared__ float ss[256], sq[256];
  const _Float16* x = X + (size_t)blockIdx.x * perSample;
  float s = 0.f, q = 0.f;
  for (int i = threadIdx.x; i < perSample; i += 256) {
    const float v = (float)x[i];
    s += v; q += v * v;
  }
  ss[threadIdx.x] = s; sq[threadIdx.x] = q; __syncthreads();
  for (int st = 128; st > 0; st >>= 1) {
    if ((int)threadIdx.x < st) { ss[threadIdx.x] += ss[threadIdx.x + st];
                                 sq[threadIdx.x] += sq[threadIdx.x + st]; }
    __syncthreads();
  }
  if (threadIdx.x == 0) {
    const float mu  = ss[0] / (float)perSample;
    const float var = sq[0] / (float)perSample - mu * mu;
    stats[blockIdx.x * 2]     = mu;
    stats[blockIdx.x * 2 + 1] = rsqrtf(var + 1e-5f);
  }
}

__global__ void gn_apply_kernel(_Float16* __restrict__ X, const float* __restrict__ stats,
                                const float* __restrict__ g, const float* __restrict__ bta,
                                int perSample, int C, int total)
{
  const int i = blockIdx.x * blockDim.x + threadIdx.x;
  if (i >= total) return;
  const int s = i / perSample;
  const int c = i % C;
  const float v = ((float)X[i] - stats[s * 2]) * stats[s * 2 + 1] * g[c] + bta[c];
  X[i] = (_Float16)LRELU(v);
}

// --------- kNN (top-10 nearest) + fused distance-softmax weights ------------
__global__ void knn_kernel(const float* __restrict__ Q, const float* __restrict__ Kp,
                           int* __restrict__ idx, float* __restrict__ w,
                           int N, int BN)
{
  const int t = blockIdx.x * blockDim.x + threadIdx.x;
  if (t >= BN) return;
  const int b = t / N;
  const float* q = Q + (size_t)t * 3;
  const float qx = q[0], qy = q[1], qz = q[2];
  const float* kp = Kp + (size_t)b * N * 3;
  float bd[10]; int bi[10];
#pragma unroll
  for (int m = 0; m < 10; ++m) { bd[m] = 3.4e38f; bi[m] = 0; }
  for (int p = 0; p < N; ++p) {
    const float dx = qx - kp[3 * p], dy = qy - kp[3 * p + 1], dz = qz - kp[3 * p + 2];
    const float d = dx * dx + dy * dy + dz * dz;
    if (d < bd[9]) {
      bd[9] = d; bi[9] = p;
#pragma unroll
      for (int m = 9; m > 0; --m) {
        if (bd[m] < bd[m - 1]) {
          const float td = bd[m]; bd[m] = bd[m - 1]; bd[m - 1] = td;
          const int  ti = bi[m]; bi[m] = bi[m - 1]; bi[m - 1] = ti;
        }
      }
    }
  }
  const float mx = -bd[0];
  float e[10], sum = 0.f;
#pragma unroll
  for (int m = 0; m < 10; ++m) { e[m] = __expf(-bd[m] - mx); sum += e[m]; }
  const float inv = 1.f / sum;
#pragma unroll
  for (int m = 0; m < 10; ++m) {
    idx[(size_t)t * 10 + m] = bi[m];
    w[(size_t)t * 10 + m]   = e[m] * inv;
  }
}

// ----------- cost-volume concat builders (gather), Kp = 288 padded ----------
__global__ void build_cat1_kernel(const _Float16* __restrict__ f1, const _Float16* __restrict__ f2,
                                  const float* __restrict__ p1, const float* __restrict__ p2,
                                  const int* __restrict__ idx, _Float16* __restrict__ cat,
                                  int N, int BN)
{
  const long long i = (long long)blockIdx.x * blockDim.x + threadIdx.x;
  const long long total = (long long)BN * 10 * 288;
  if (i >= total) return;
  const int c = (int)(i % 288);
  const long long rm = i / 288;
  const int m = (int)(rm % 10);
  const int r = (int)(rm / 10);                // r = b*N + n
  const int b = r / N;
  const int jn = idx[(size_t)r * 10 + m];
  const size_t jr = (size_t)b * N + jn;
  _Float16 v;
  if (c < 128)       v = f1[(size_t)r * 128 + c];
  else if (c < 256)  v = f2[jr * 128 + (c - 128)];
  else if (c < 259)  v = (_Float16)(p2[jr * 3 + (c - 256)] - p1[(size_t)r * 3 + (c - 256)]);
  else               v = (_Float16)0.f;
  cat[i] = v;
}

__global__ void build_cat2_kernel(const _Float16* __restrict__ pc, const float* __restrict__ p1,
                                  const int* __restrict__ idx, _Float16* __restrict__ cat,
                                  int N, int BN)
{
  const long long i = (long long)blockIdx.x * blockDim.x + threadIdx.x;
  const long long total = (long long)BN * 10 * 288;
  if (i >= total) return;
  const int c = (int)(i % 288);
  const long long rm = i / 288;
  const int m = (int)(rm % 10);
  const int r = (int)(rm / 10);
  const int b = r / N;
  const int jn = idx[(size_t)r * 10 + m];
  const size_t jr = (size_t)b * N + jn;
  _Float16 v;
  if (c < 256)      v = pc[jr * 256 + c];
  else if (c < 259) v = (_Float16)(p1[jr * 3 + (c - 256)] - p1[(size_t)r * 3 + (c - 256)]);
  else              v = (_Float16)0.f;
  cat[i] = v;
}

// --------------- neighbor-weighted sum over M=10 (softmax weights) ----------
__global__ void wsum_kernel(const _Float16* __restrict__ h, const float* __restrict__ w,
                            _Float16* __restrict__ out, int BN, int C)
{
  const int t = blockIdx.x * blockDim.x + threadIdx.x;
  if (t >= BN * C) return;
  const int r = t / C, c = t - r * C;
  float s = 0.f;
#pragma unroll
  for (int m = 0; m < 10; ++m)
    s += w[(size_t)r * 10 + m] * (float)h[((size_t)r * 10 + m) * C + c];
  out[t] = (_Float16)s;
}

// ------------------------- mask head concat [f1 | cv] -----------------------
__global__ void build_catmk_kernel(const _Float16* __restrict__ f1, const _Float16* __restrict__ cv,
                                   _Float16* __restrict__ cat, int BN)
{
  const int i = blockIdx.x * blockDim.x + threadIdx.x;
  if (i >= BN * 384) return;
  const int r = i / 384, c = i - r * 384;
  cat[i] = (c < 128) ? f1[(size_t)r * 128 + c] : cv[(size_t)r * 256 + (c - 128)];
}

// --------- softmax over points (axis=2) per (b,c) and emb = cv * mask -------
__global__ void mask_emb_kernel(const _Float16* __restrict__ mk, const _Float16* __restrict__ cv,
                                _Float16* __restrict__ emb, int N)
{
  const int b = blockIdx.x >> 8;
  const int c = blockIdx.x & 255;
  __shared__ float red[256];
  const int tid = threadIdx.x;
  float mx = -3.4e38f;
  for (int n = tid; n < N; n += 256)
    mx = fmaxf(mx, (float)mk[((size_t)b * N + n) * 256 + c]);
  red[tid] = mx; __syncthreads();
  for (int s = 128; s > 0; s >>= 1) {
    if (tid < s) red[tid] = fmaxf(red[tid], red[tid + s]);
    __syncthreads();
  }
  mx = red[0]; __syncthreads();
  float sum = 0.f;
  for (int n = tid; n < N; n += 256)
    sum += __expf((float)mk[((size_t)b * N + n) * 256 + c] - mx);
  red[tid] = sum; __syncthreads();
  for (int s = 128; s > 0; s >>= 1) {
    if (tid < s) red[tid] += red[tid + s];
    __syncthreads();
  }
  const float inv = 1.f / red[0];
  for (int n = tid; n < N; n += 256) {
    const size_t o = ((size_t)b * N + n) * 256 + c;
    emb[o] = (_Float16)((float)cv[o] * __expf((float)mk[o] - mx) * inv);
  }
}

// ------- decoder tail: 64->2, tanh elevation, softmax over M, spherical -----
__global__ void final_kernel(const _Float16* __restrict__ dh2, const float* __restrict__ w3,
                             const float* __restrict__ b3, const float* __restrict__ rtheta,
                             float* __restrict__ out, int B, int nout)
{
  const int t = blockIdx.x * blockDim.x + threadIdx.x;
  if (t >= B * nout) return;
  const int b = t / nout, j = t - b * nout;
  const int N = nout * 10;
  float o0[10], o1[10];
  float mx = -3.4e38f;
#pragma unroll
  for (int m = 0; m < 10; ++m) {
    const _Float16* h = dh2 + ((size_t)b * N + (size_t)m * nout + j) * 64;
    float s0 = b3[0], s1 = b3[1];
    for (int k = 0; k < 64; ++k) {
      const float hv = (float)h[k];
      s0 += w3[k] * hv; s1 += w3[64 + k] * hv;
    }
    o0[m] = s0; o1[m] = s1; mx = fmaxf(mx, s1);
  }
  float e[10], sum = 0.f;
#pragma unroll
  for (int m = 0; m < 10; ++m) { e[m] = __expf(o1[m] - mx); sum += e[m]; }
  const float inv = 1.f / sum;
  float phi = 0.f, r = 0.f, th = 0.f;
#pragma unroll
  for (int m = 0; m < 10; ++m) {
    const float wgt = e[m] * inv;
    phi += wgt * tanhf(o0[m]) * 1.5707963267948966f;
    const float* rt = rtheta + ((size_t)b * N + (size_t)m * nout + j) * 2;
    r  += wgt * rt[0];
    th += wgt * rt[1];
  }
  const float cp = cosf(phi), sp = sinf(phi);
  float* pts = out + (size_t)t * 3;            // points_out1 (B,nout,3)
  pts[0] = r * cp * cosf(th);
  pts[1] = r * cp * sinf(th);
  pts[2] = r * sp;
  out[(size_t)B * nout * 3 + t] = phi;         // phi1_pre (B,nout)
}

// ---------------------------------------------------------------------------
extern "C" void kernel_launch(void* const* d_in, const int* in_sizes, int n_in,
                              void* d_out, int out_size, void* d_ws, size_t ws_size,
                              hipStream_t stream) {
  (void)n_in; (void)out_size; (void)ws_size;

  const float* xyz1   = (const float*)d_in[0];
  const float* feat1  = (const float*)d_in[1];
  const float* xyz2   = (const float*)d_in[2];
  const float* feat2  = (const float*)d_in[3];
  // d_in[4] = nout (scalar; derived from sizes instead)
  const float* rtheta = (const float*)d_in[5];
  const float* enc_w1 = (const float*)d_in[6];
  const float* enc_b1 = (const float*)d_in[7];
  const float* gn1_g  = (const float*)d_in[8];
  const float* gn1_b  = (const float*)d_in[9];
  const float* enc_w2 = (const float*)d_in[10];
  const float* enc_b2 = (const float*)d_in[11];
  const float* gn2_g  = (const float*)d_in[12];
  const float* gn2_b  = (const float*)d_in[13];
  const float* enc_w3 = (const float*)d_in[14];
  const float* enc_b3 = (const float*)d_in[15];
  const float* cv1_w1 = (const float*)d_in[16];
  const float* cv1_b1 = (const float*)d_in[17];
  const float* cv1_w2 = (const float*)d_in[18];
  const float* cv1_b2 = (const float*)d_in[19];
  const float* cv1_w3 = (const float*)d_in[20];
  const float* cv1_b3 = (const float*)d_in[21];
  const float* cv2_w1 = (const float*)d_in[22];
  const float* cv2_b1 = (const float*)d_in[23];
  const float* cv2_w2 = (const float*)d_in[24];
  const float* cv2_b2 = (const float*)d_in[25];
  const float* cv2_w3 = (const float*)d_in[26];
  const float* cv2_b3 = (const float*)d_in[27];
  const float* mk_w1  = (const float*)d_in[28];
  const float* mk_b1  = (const float*)d_in[29];
  const float* mk_w2  = (const float*)d_in[30];
  const float* mk_b2  = (const float*)d_in[31];
  const float* dec_w1 = (const float*)d_in[32];
  const float* dec_b1 = (const float*)d_in[33];
  const float* dec_w2 = (const float*)d_in[34];
  const float* dec_b2 = (const float*)d_in[35];
  const float* dec_w3 = (const float*)d_in[36];
  const float* dec_b3 = (const float*)d_in[37];

  const int BN   = in_sizes[0] / 3;     // B*N (10240): multiple of 64
  const int B    = 2;
  const int N    = BN / B;
  const int nout = N / 10;
  const int R    = BN * 10;             // cost-volume rows (102400): mult of 64

  // ---- bump workspace allocator (256B-aligned chunks) ----
  char* base = (char*)d_ws;
  size_t off = 0;
  auto alloc = [&](size_t bytes) -> void* {
    void* p = base + off;
    off = (off + bytes + 255) & ~(size_t)255;
    return p;
  };
  typedef _Float16 h16;
  h16*  f1buf  = (h16*)alloc((size_t)BN * 128 * 2);
  h16*  f2buf  = (h16*)alloc((size_t)BN * 128 * 2);
  float* stats = (float*)alloc(16 * 4);
  int*  idx1   = (int*)alloc((size_t)BN * 10 * 4);
  float* wkn1  = (float*)alloc((size_t)BN * 10 * 4);
  int*  idx2   = (int*)alloc((size_t)BN * 10 * 4);
  float* wkn2  = (float*)alloc((size_t)BN * 10 * 4);
  h16*  pc     = (h16*)alloc((size_t)BN * 256 * 2);
  h16*  cvbuf  = (h16*)alloc((size_t)BN * 256 * 2);
  h16*  bufH1  = (h16*)alloc((size_t)R * 128 * 2);
  h16*  bufH2  = (h16*)alloc((size_t)R * 128 * 2);
  h16*  bufH3  = (h16*)alloc((size_t)R * 256 * 2);
  h16*  bufCat = (h16*)alloc((size_t)R * 288 * 2);
  // f16 weights (zero-padded in K)
  h16* enc_w2f = (h16*)alloc(128 * 64 * 2);
  h16* enc_w3f = (h16*)alloc(128 * 128 * 2);
  h16* cv1_w1f = (h16*)alloc(128 * 288 * 2);
  h16* cv1_w2f = (h16*)alloc(128 * 128 * 2);
  h16* cv1_w3f = (h16*)alloc(256 * 128 * 2);
  h16* cv2_w1f = (h16*)alloc(128 * 288 * 2);
  h16* cv2_w2f = (h16*)alloc(128 * 128 * 2);
  h16* cv2_w3f = (h16*)alloc(256 * 128 * 2);
  h16* mk_w1f  = (h16*)alloc(256 * 384 * 2);
  h16* mk_w2f  = (h16*)alloc(256 * 256 * 2);
  h16* dec_w1f = (h16*)alloc(128 * 256 * 2);
  h16* dec_w2f = (h16*)alloc(64 * 128 * 2);
  // late-stage buffers overlay bufCat (dead after cv2's first GEMM reads it)
  h16* catmk = bufCat;
  h16* mk1   = catmk + (size_t)BN * 384;
  h16* mk2   = mk1   + (size_t)BN * 256;
  h16* embb  = mk2   + (size_t)BN * 256;
  h16* dh1   = embb  + (size_t)BN * 256;
  h16* dh2   = dh1   + (size_t)BN * 128;

  auto cdiv = [](long long a, int b) { return (unsigned)((a + b - 1) / b); };
  auto wcvt = [&](const float* w, h16* o, int O, int K, int Kp) {
    wcvt_kernel<<<cdiv((long long)O * Kp, 256), 256, 0, stream>>>(w, o, O, K, Kp);
  };
  auto gemm = [&](const h16* A, const h16* Wm, const float* bi, h16* Cc,
                  int rows, int K, int O, int act) {
    const int nTiles = (rows >> 6) * (O >> 4);
    wmma_gemm_kernel<<<(nTiles + 7) / 8, 256, 0, stream>>>(A, Wm, bi, Cc, rows, K, O, act);
  };

  // -------------------- weight conversion (tiny, once per launch) ------
  wcvt(enc_w2, enc_w2f, 128, 64, 64);
  wcvt(enc_w3, enc_w3f, 128, 128, 128);
  wcvt(cv1_w1, cv1_w1f, 128, 259, 288);
  wcvt(cv1_w2, cv1_w2f, 128, 128, 128);
  wcvt(cv1_w3, cv1_w3f, 256, 128, 128);
  wcvt(cv2_w1, cv2_w1f, 128, 259, 288);
  wcvt(cv2_w2, cv2_w2f, 128, 128, 128);
  wcvt(cv2_w3, cv2_w3f, 256, 128, 128);
  wcvt(mk_w1,  mk_w1f,  256, 384, 384);
  wcvt(mk_w2,  mk_w2f,  256, 256, 256);
  wcvt(dec_w1, dec_w1f, 128, 256, 256);
  wcvt(dec_w2, dec_w2f, 64, 128, 128);

  // -------------------- encoder (both streams) --------------------
  for (int s = 0; s < 2; ++s) {
    const float* xs = s ? feat2 : feat1;
    h16* fs = s ? f2buf : f1buf;
    enc1_kernel<<<cdiv((long long)BN * 64, 256), 256, 0, stream>>>(xs, enc_w1, enc_b1, bufH1, BN);
    gn_stats_kernel<<<B, 256, 0, stream>>>(bufH1, stats, N * 64);
    gn_apply_kernel<<<cdiv((long long)BN * 64, 256), 256, 0, stream>>>(
        bufH1, stats, gn1_g, gn1_b, N * 64, 64, BN * 64);
    gemm(bufH1, enc_w2f, enc_b2, bufH2, BN, 64, 128, 0);
    gn_stats_kernel<<<B, 256, 0, stream>>>(bufH2, stats, N * 128);
    gn_apply_kernel<<<cdiv((long long)BN * 128, 256), 256, 0, stream>>>(
        bufH2, stats, gn2_g, gn2_b, N * 128, 128, BN * 128);
    gemm(bufH2, enc_w3f, enc_b3, fs, BN, 128, 128, 0);
  }

  // -------------------- cost volume: point-to-patch --------------------
  knn_kernel<<<cdiv(BN, 256), 256, 0, stream>>>(xyz1, xyz2, idx1, wkn1, N, BN);
  build_cat1_kernel<<<cdiv((long long)R * 288, 256), 256, 0, stream>>>(
      f1buf, f2buf, xyz1, xyz2, idx1, bufCat, N, BN);
  gemm(bufCat, cv1_w1f, cv1_b1, bufH1, R, 288, 128, 1);
  gemm(bufH1,  cv1_w2f, cv1_b2, bufH2, R, 128, 128, 1);
  gemm(bufH2,  cv1_w3f, cv1_b3, bufH3, R, 128, 256, 0);
  wsum_kernel<<<cdiv((long long)BN * 256, 256), 256, 0, stream>>>(bufH3, wkn1, pc, BN, 256);

  // -------------------- cost volume: patch-to-patch --------------------
  knn_kernel<<<cdiv(BN, 256), 256, 0, stream>>>(xyz1, xyz1, idx2, wkn2, N, BN);
  build_cat2_kernel<<<cdiv((long long)R * 288, 256), 256, 0, stream>>>(
      pc, xyz1, idx2, bufCat, N, BN);
  gemm(bufCat, cv2_w1f, cv2_b1, bufH1, R, 288, 128, 1);
  gemm(bufH1,  cv2_w2f, cv2_b2, bufH2, R, 128, 128, 1);
  gemm(bufH2,  cv2_w3f, cv2_b3, bufH3, R, 128, 256, 0);
  wsum_kernel<<<cdiv((long long)BN * 256, 256), 256, 0, stream>>>(bufH3, wkn2, cvbuf, BN, 256);

  // -------------------- mask head + softmax over points ----------------
  build_catmk_kernel<<<cdiv((long long)BN * 384, 256), 256, 0, stream>>>(f1buf, cvbuf, catmk, BN);
  gemm(catmk, mk_w1f, mk_b1, mk1, BN, 384, 256, 1);
  gemm(mk1,   mk_w2f, mk_b2, mk2, BN, 256, 256, 0);
  mask_emb_kernel<<<B * 256, 256, 0, stream>>>(mk2, cvbuf, embb, N);

  // -------------------- decoder + spherical reconstruction -------------
  gemm(embb, dec_w1f, dec_b1, dh1, BN, 256, 128, 1);
  gemm(dh1,  dec_w2f, dec_b2, dh2, BN, 128, 64, 1);
  final_kernel<<<cdiv(B * nout, 256), 256, 0, stream>>>(
      dh2, dec_w3, dec_b3, rtheta, (float*)d_out, B, nout);
}
